// ReactionDiffusionPath_73555609911403
// MI455X (gfx1250) — matrix operand order repaired and checked by reading, not compiled, not run on hardware
//
#include <hip/hip_runtime.h>
#include <hip/hip_bf16.h>

typedef __attribute__((ext_vector_type(16))) __bf16 bf16x16;
typedef __attribute__((ext_vector_type(8)))  float  v8f;
typedef __attribute__((ext_vector_type(4)))  float  v4f;

#define BATCH  16
#define SEQ    4096
#define DIM    256
#define HW     64          // 64x64 spatial grid
#define STEPS  10
#define DT_C   0.3f
#define PW     66          // padded width (zero border)

// ---------------------------------------------------------------------------
// Kernel 0: convert W (f32, [e,d] row-major, d contiguous) to bf16 once.
// ---------------------------------------------------------------------------
__global__ __launch_bounds__(256) void convert_w_kernel(
    const float* __restrict__ W, __bf16* __restrict__ Wb) {
  int i = blockIdx.x * 256 + threadIdx.x;   // DIM*DIM = 65536 elements
  Wb[i] = (__bf16)W[i];
}

// ---------------------------------------------------------------------------
// Kernel 1: projected[b,s,e] = sum_d x[b,s,d] * W[e,d]  (+ bias[e])
// written transposed as field[b,e,s] so kernel 2 reads are coalesced.
// Each wave computes a 16(M) x 64(N) strip with v_wmma_f32_16x16x32_bf16.
// blockDim = (32, 8): 8 waves per block. 2048 blocks => 16384 waves
//   gw = global wave id;  m_tile = gw>>2 (rows m0..m0+15), n0 = (gw&3)*64.
// ---------------------------------------------------------------------------
__global__ __launch_bounds__(256) void gemm_bf16_kernel(
    const float* __restrict__ X,        // [65536, 256] f32 (b*4096+s, d)
    const __bf16* __restrict__ Wb,      // [256, 256] bf16 (e, d)
    const float* __restrict__ bias,     // [256]
    float* __restrict__ field) {        // [16, 256, 4096] f32 (b, e, s)
  const int lane = threadIdx.x;                 // 0..31 (wave32)
  const int gw   = blockIdx.x * 8 + threadIdx.y;
  const int m0   = (gw >> 2) * 16;
  const int n0   = (gw & 3) * 64;

  const int row    = m0 + (lane & 15);          // A-matrix row (M)
  const int koffA  = (lane < 16) ? 0 : 16;      // A K-chunk base within 32
  const int koffB  = (lane < 16) ? 0 : 16;      // B K-chunk base within 32

  v8f acc0 = {}, acc1 = {}, acc2 = {}, acc3 = {};
  const float* xrow = X + (size_t)row * DIM;

  for (int k0 = 0; k0 < DIM; k0 += 32) {
    // ---- A fragment: f32 loads (2 x 32B) + convert to bf16 ----
    v8f a_lo = *(const v8f*)(xrow + k0 + koffA);        // K = koffA..koffA+7
    v8f a_hi = *(const v8f*)(xrow + k0 + koffA + 8);    // K = koffA+8..+15
    bf16x16 afrag;
#pragma unroll
    for (int i = 0; i < 8; ++i) {
      afrag[i]     = (__bf16)a_lo[i];
      afrag[8 + i] = (__bf16)a_hi[i];
    }
    // ---- B fragments: 16 consecutive bf16 per lane (32B) ----
    const __bf16* wbase = Wb + (size_t)(n0 + (lane & 15)) * DIM + k0 + koffB;
    bf16x16 b0 = *(const bf16x16*)(wbase);
    bf16x16 b1 = *(const bf16x16*)(wbase + 16 * DIM);
    bf16x16 b2 = *(const bf16x16*)(wbase + 32 * DIM);
    bf16x16 b3 = *(const bf16x16*)(wbase + 48 * DIM);

    acc0 = __builtin_amdgcn_wmma_f32_16x16x32_bf16(false, afrag, false, b0,
                                                   (short)0, acc0, false, false);
    acc1 = __builtin_amdgcn_wmma_f32_16x16x32_bf16(false, afrag, false, b1,
                                                   (short)0, acc1, false, false);
    acc2 = __builtin_amdgcn_wmma_f32_16x16x32_bf16(false, afrag, false, b2,
                                                   (short)0, acc2, false, false);
    acc3 = __builtin_amdgcn_wmma_f32_16x16x32_bf16(false, afrag, false, b3,
                                                   (short)0, acc3, false, false);
  }

  // ---- store: C layout lane L holds N = (L&15), M = i + 8*(L>=16) for VGPR i
  const int b_idx  = m0 >> 12;                       // 4096 rows per batch
  const int s_base = (m0 & 4095) + ((lane >> 4) << 3);
  v8f accs[4] = {acc0, acc1, acc2, acc3};
#pragma unroll
  for (int t = 0; t < 4; ++t) {
    const int e  = n0 + t * 16 + (lane & 15);
    const float bv = bias[e];
    float* dst = field + (((size_t)b_idx * DIM + e) << 12) + s_base;
    v4f lo = {accs[t][0] + bv, accs[t][1] + bv, accs[t][2] + bv, accs[t][3] + bv};
    v4f hi = {accs[t][4] + bv, accs[t][5] + bv, accs[t][6] + bv, accs[t][7] + bv};
    *(v4f*)(dst)     = lo;     // 16B aligned: s_base is a multiple of 8
    *(v4f*)(dst + 4) = hi;
  }
}

// ---------------------------------------------------------------------------
// fast tanh: (e^{2x}-1)/(e^{2x}+1) via v_exp_f32 + v_rcp_f32 (trans units)
// ---------------------------------------------------------------------------
__device__ __forceinline__ float tanh_fast(float x) {
  x = fminf(fmaxf(x, -15.0f), 15.0f);
  float t = __builtin_amdgcn_exp2f(x * 2.8853900817779268f);  // e^{2x}
  return (t - 1.0f) * __builtin_amdgcn_rcpf(t + 1.0f);
}

// ---------------------------------------------------------------------------
// Kernel 2: per-(b,e) plane reaction-diffusion entirely in LDS.
// One block per plane (4096 blocks, 256 threads). 66x66 zero-padded
// ping-pong buffers; borders stay zero (matches zero-pad Laplacian).
// Ends with block mean reduction -> out[b*256+e].
// ---------------------------------------------------------------------------
__global__ __launch_bounds__(256) void evolve_kernel(
    const float* __restrict__ field,    // [16, 256, 4096]
    const float* __restrict__ dcoef,    // [1]
    float* __restrict__ out) {          // [16*256]
  __shared__ float buf[2][PW * PW];
  const int t     = threadIdx.x;
  const int plane = blockIdx.x;                 // b*256 + e
  const float Dc  = dcoef[0];

  // zero both buffers (borders must be zero every step)
  for (int i = t; i < PW * PW; i += 256) { buf[0][i] = 0.0f; buf[1][i] = 0.0f; }
  __syncthreads();

  // coalesced load of the plane into interior of buf[0]
  const float* src = field + ((size_t)plane << 12);
#pragma unroll
  for (int j = 0; j < 16; ++j) {
    int i = t + j * 256;
    int h = i >> 6, w = i & 63;
    buf[0][(h + 1) * PW + (w + 1)] = src[i];
  }
  __syncthreads();

  int cur = 0;
  for (int step = 0; step < STEPS; ++step) {
    const float* __restrict__ c0 = buf[cur];
    float* __restrict__ c1 = buf[1 - cur];
#pragma unroll
    for (int j = 0; j < 16; ++j) {
      int i = t + j * 256;
      int h = i >> 6, w = i & 63;
      int p = (h + 1) * PW + (w + 1);
      float c   = c0[p];
      float lap = c0[p - 1] + c0[p + 1] + c0[p - PW] + c0[p + PW] - 4.0f * c;
      c1[p] = c + DT_C * (Dc * lap + tanh_fast(c) - c);
    }
    __syncthreads();
    cur ^= 1;
  }

  // spatial mean: per-thread partial over its 16 cells, then tree reduce
  float sum = 0.0f;
#pragma unroll
  for (int j = 0; j < 16; ++j) {
    int i = t + j * 256;
    int h = i >> 6, w = i & 63;
    sum += buf[cur][(h + 1) * PW + (w + 1)];
  }
  __syncthreads();                 // all reads done before reusing buf[0]
  buf[0][t] = sum;
  __syncthreads();
  for (int s = 128; s > 0; s >>= 1) {
    if (t < s) buf[0][t] += buf[0][t + s];
    __syncthreads();
  }
  if (t == 0) out[plane] = buf[0][0] * (1.0f / 4096.0f);
}

// ---------------------------------------------------------------------------
extern "C" void kernel_launch(void* const* d_in, const int* in_sizes, int n_in,
                              void* d_out, int out_size, void* d_ws, size_t ws_size,
                              hipStream_t stream) {
  const float* x     = (const float*)d_in[0];   // [16,4096,256]
  const float* W     = (const float*)d_in[1];   // [256,256]
  const float* bias  = (const float*)d_in[2];   // [256]
  const float* dcoef = (const float*)d_in[3];   // [1]
  float* out = (float*)d_out;                   // [16,256]

  // workspace: field f32 [16,256,4096] = 64MB, then W in bf16 (128KB)
  float*  field = (float*)d_ws;
  __bf16* Wb    = (__bf16*)((char*)d_ws + (size_t)BATCH * DIM * SEQ * sizeof(float));

  convert_w_kernel<<<dim3(DIM * DIM / 256), dim3(256), 0, stream>>>(W, Wb);
  gemm_bf16_kernel<<<dim3(2048), dim3(32, 8), 0, stream>>>(x, Wb, bias, field);
  evolve_kernel<<<dim3(BATCH * DIM), dim3(256), 0, stream>>>(field, dcoef, out);
}
